// GATSegmentationModel_31155692765918
// MI455X (gfx1250) — matrix operand order repaired and checked by reading, not compiled, not run on hardware
//
#include <hip/hip_runtime.h>

// ---------------------------------------------------------------------------
// Types for WMMA fragments (gfx1250, wave32)
// ---------------------------------------------------------------------------
typedef __attribute__((ext_vector_type(16))) __bf16        v16bf;
typedef __attribute__((ext_vector_type(8)))  float         v8f;
typedef __attribute__((ext_vector_type(8)))  unsigned short us8;

union ABfrag { v16bf v; us8 u[2]; };

__device__ __forceinline__ unsigned short f2bf(float f) {
  union { float f; unsigned u; } x; x.f = f;
  unsigned r = x.u + 0x7fffu + ((x.u >> 16) & 1u);   // round-to-nearest-even
  return (unsigned short)(r >> 16);
}

// ---------------------------------------------------------------------------
// bf16 conversion / transpose helpers
// ---------------------------------------------------------------------------
__global__ void k_cvt_bf(const float* __restrict__ in, unsigned short* __restrict__ out, int n) {
  int i = blockIdx.x * blockDim.x + threadIdx.x;
  if (i < n) out[i] = f2bf(in[i]);
}

// out[c][r] = in[r][c]; in is [R][C], out is [C][R]
__global__ void k_transpose_cvt(const float* __restrict__ in, unsigned short* __restrict__ out,
                                int R, int C) {
  int i = blockIdx.x * blockDim.x + threadIdx.x;
  if (i >= R * C) return;
  int r = i / C, c = i % C;
  out[(size_t)c * R + r] = f2bf(in[i]);
}

// ---------------------------------------------------------------------------
// GEMM: C[M][N] (f32, +bias, optional ReLU) = A[M][K](bf16) * Bt[N][K](bf16)^T
// One wave computes a 16 x (16*NBLK) tile: one A fragment is reused by NBLK
// B fragments / accumulators per k-step (v_wmma_f32_16x16x32_bf16).
// A-fragment per ISA: lane L holds row m=L%16; halves 0..7 -> K=kb+(L/16)*8+0..7,
//                     halves 8..15 -> K=kb+16+(L/16)*8+0..7  (two 16B loads).
// B-fragment: lane L holds col n=L%16; same K mapping as A
//                     (two 16B loads from the Bt row).
// ---------------------------------------------------------------------------
template <int NBLK>
__global__ void k_gemm_bf16(const unsigned short* __restrict__ A,
                            const unsigned short* __restrict__ Bt,
                            const float* __restrict__ bias,
                            float* __restrict__ C,
                            int M, int N, int K, int relu) {
  int lane = threadIdx.x & 31;
  int wave = threadIdx.x >> 5;
  int tile = blockIdx.x * (blockDim.x >> 5) + wave;
  int mt = M >> 4;                 // 16-row tiles
  int ntb = N / (16 * NBLK);       // (16*NBLK)-col tiles
  if (tile >= mt * ntb) return;
  int tm  = tile % mt;
  int tn0 = (tile / mt) * NBLK;
  int l16 = lane & 15;
  int hi  = lane >> 4;

  const unsigned short* arow = A  + (size_t)(tm * 16 + l16) * K + hi * 8;
  const unsigned short* brow = Bt + (size_t)(tn0 * 16 + l16) * K + hi * 8;
  const size_t bstep = (size_t)16 * K;           // next 16-col B tile

  v8f acc[NBLK];
#pragma unroll
  for (int t = 0; t < NBLK; ++t) acc[t] = {};

  for (int kb = 0; kb < K; kb += 32) {
    ABfrag a;
    a.u[0] = *(const us8*)(arow + kb);
    a.u[1] = *(const us8*)(arow + kb + 16);
    ABfrag b[NBLK];
#pragma unroll
    for (int t = 0; t < NBLK; ++t) {
      b[t].u[0] = *(const us8*)(brow + t * bstep + kb);
      b[t].u[1] = *(const us8*)(brow + t * bstep + kb + 16);
    }
    if (kb + 32 < K) {
      __builtin_prefetch(arow + kb + 32, 0, 3);   // global_prefetch_b8, near-cache
      __builtin_prefetch(brow + kb + 32, 0, 3);
    }
#pragma unroll
    for (int t = 0; t < NBLK; ++t)
      acc[t] = __builtin_amdgcn_wmma_f32_16x16x32_bf16(false, a.v, false, b[t].v,
                                                       (short)0, acc[t], false, false);
  }

  // C/D layout: VGPR v, lane L -> row = v + (L/16)*8, col = L%16
  int orow0 = tm * 16 + hi * 8;
#pragma unroll
  for (int t = 0; t < NBLK; ++t) {
    int ocol = (tn0 + t) * 16 + l16;
    float bv = bias ? bias[ocol] : 0.f;
#pragma unroll
    for (int r = 0; r < 8; ++r) {
      float o = acc[t][r] + bv;
      if (relu) o = fmaxf(o, 0.f);
      C[(size_t)(orow0 + r) * N + ocol] = o;
    }
  }
}

// ---------------------------------------------------------------------------
// Squeeze-excite pieces
// ---------------------------------------------------------------------------
__global__ void k_col_mean(const float* __restrict__ Y, float* __restrict__ mean, int M, int N) {
  __shared__ float sm[256];
  int n = blockIdx.x;
  float s = 0.f;
  for (int m = threadIdx.x; m < M; m += blockDim.x) s += Y[(size_t)m * N + n];
  sm[threadIdx.x] = s;
  __syncthreads();
  for (int st = 128; st > 0; st >>= 1) {
    if ((int)threadIdx.x < st) sm[threadIdx.x] += sm[threadIdx.x + st];
    __syncthreads();
  }
  if (threadIdx.x == 0) mean[n] = sm[0] / (float)M;
}

// gate[n] = sigmoid(w2[n][:] . relu(w1 @ mean + b1) + b2[n]);  single block
__global__ void k_se_mlp(const float* __restrict__ mean,
                         const float* __restrict__ w1, const float* __restrict__ b1,
                         const float* __restrict__ w2, const float* __restrict__ b2,
                         float* __restrict__ gate, int Cc, int Hh) {
  __shared__ float hid[32];
  int t = threadIdx.x;
  if (t < Hh) {
    float s = b1[t];
    for (int c = 0; c < Cc; ++c) s += mean[c] * w1[t * Cc + c];
    hid[t] = fmaxf(s, 0.f);
  }
  __syncthreads();
  for (int n = t; n < Cc; n += blockDim.x) {
    float s = b2[n];
    for (int j = 0; j < Hh; ++j) s += hid[j] * w2[n * Hh + j];
    gate[n] = 1.f / (1.f + __expf(-s));
  }
}

// Xbf[m][n] = bf16(Y[m][n] * gate[n])
__global__ void k_scale_cvt(const float* __restrict__ Y, const float* __restrict__ gate,
                            unsigned short* __restrict__ Xbf, int M, int N) {
  int i = blockIdx.x * blockDim.x + threadIdx.x;
  if (i >= M * N) return;
  Xbf[i] = f2bf(Y[i] * gate[i % N]);
}

// Node features: flat f = c*4096 + hw  ->  Xg[f] = Y3[hw][c]*gate[c]
__global__ void k_stage3_nodes(const float* __restrict__ Y3, const float* __restrict__ gate,
                               unsigned short* __restrict__ Xg, int total) {
  int f = blockIdx.x * blockDim.x + threadIdx.x;
  if (f >= total) return;
  int c = f >> 12;       // /4096
  int hw = f & 4095;
  Xg[f] = f2bf(Y3[(size_t)hw * 64 + c] * gate[c]);
}

// ---------------------------------------------------------------------------
// GAT per-node attention scores: as[m][hd] = h[m,hd,:].a_src[hd], ad likewise
// ---------------------------------------------------------------------------
template <int COUT>
__global__ void k_gat_scores(const float* __restrict__ h,
                             const float* __restrict__ a_src, const float* __restrict__ a_dst,
                             float* __restrict__ as_, float* __restrict__ ad_) {
  int m = blockIdx.x * blockDim.x + threadIdx.x;
  if (m >= 8192) return;
  const int CH = 4 * COUT;
#pragma unroll
  for (int hd = 0; hd < 4; ++hd) {
    float s = 0.f, d = 0.f;
#pragma unroll
    for (int c = 0; c < COUT; ++c) {
      float hv = h[(size_t)m * CH + hd * COUT + c];
      s += hv * a_src[hd * COUT + c];
      d += hv * a_dst[hd * COUT + c];
    }
    as_[m * 4 + hd] = s;
    ad_[m * 4 + hd] = d;
  }
}

// ---------------------------------------------------------------------------
// GAT softmax-aggregation, gather form, online softmax (deterministic, no atomics).
// Reference edge list structure: for grid node v (v<4096, 64x64 grid), every
// in-bounds radius-5 neighbor contributes exactly TWO directed edges into v
// (cat([e, e.flip(0)]) over the full ordered-offset list), plus ONE self loop.
// Rows 4096..8191 carry only the self loop (alpha == 1).
// One wave per destination node; lane p covers channel ch=lane+32p, head ch/COUT.
// ---------------------------------------------------------------------------
template <int COUT>
__global__ void k_gat_aggregate(const float* __restrict__ h,      // [8192][4*COUT]
                                const float* __restrict__ as_,    // [8192][4]
                                const float* __restrict__ ad_,    // [8192][4]
                                const float* __restrict__ bias,   // [4*COUT]
                                float* __restrict__ out,          // [8192][4*COUT] (ReLU'd)
                                unsigned short* __restrict__ out_bf) { // optional bf16 copy
  constexpr int CH = 4 * COUT;
  constexpr int P  = (CH + 31) / 32;
  int lane = threadIdx.x & 31;
  int wave = threadIdx.x >> 5;
  int v = blockIdx.x * (blockDim.x >> 5) + wave;
  if (v >= 8192) return;

  int   ch[P], hd[P];
  float m[P], l[P], acc[P], adv[P];
#pragma unroll
  for (int p = 0; p < P; ++p) {
    ch[p] = lane + 32 * p;
    hd[p] = ch[p] / COUT;
    adv[p] = ad_[v * 4 + hd[p]];
    m[p] = -1e30f; l[p] = 0.f; acc[p] = 0.f;
  }

  auto edge = [&](int u, float wgt) {
#pragma unroll
    for (int p = 0; p < P; ++p) {
      float e = as_[u * 4 + hd[p]] + adv[p];
      e = (e > 0.f) ? e : 0.2f * e;                 // LeakyReLU(0.2)
      float hv = h[(size_t)u * CH + ch[p]];
      if (e > m[p]) {                               // online softmax rescale
        float s = __expf(m[p] - e);
        l[p] *= s; acc[p] *= s; m[p] = e;
      }
      float ww = wgt * __expf(e - m[p]);
      l[p] += ww;
      acc[p] += ww * hv;
    }
  };

  if (v < 4096) {
    int i = v >> 6, j = v & 63;
    for (int dx = -5; dx <= 5; ++dx) {
      int ii = i + dx;
      if ((unsigned)ii >= 64u) continue;
      for (int dy = -5; dy <= 5; ++dy) {
        if (dx == 0 && dy == 0) continue;
        int jj = j + dy;
        if ((unsigned)jj >= 64u) continue;
        edge(ii * 64 + jj, 2.0f);                   // duplicated directed edge
      }
    }
  }
  edge(v, 1.0f);                                    // self loop

#pragma unroll
  for (int p = 0; p < P; ++p) {
    float o = acc[p] / l[p] + bias[ch[p]];
    o = fmaxf(o, 0.f);                              // ReLU after each GAT layer
    out[(size_t)v * CH + ch[p]] = o;
    if (out_bf) out_bf[(size_t)v * CH + ch[p]] = f2bf(o);
  }
}

// ---------------------------------------------------------------------------
// Host launch
// ---------------------------------------------------------------------------
static inline size_t alup(size_t x) { return (x + 255) & ~(size_t)255; }

extern "C" void kernel_launch(void* const* d_in, const int* in_sizes, int n_in,
                              void* d_out, int out_size, void* d_ws, size_t ws_size,
                              hipStream_t stream) {
  (void)in_sizes; (void)out_size; (void)ws_size;
  const float* x2_raw = (const float*)d_in[0];

  // Param leaves, JAX sorted-key order. Either one concatenated blob (n_in small)
  // or individual leaf pointers (n_in>=29).
  const float *g1W, *g1ad, *g1as, *g1b, *g2W, *g2ad, *g2as, *g2b;
  const float *rb1, *rb2, *rb3;
  const float *s1b1, *s1b2, *s1w1, *s1w2, *s2b1, *s2b2, *s2w1, *s2w2, *s3b1, *s3b2, *s3w1, *s3w2;
  const float *rw1, *rw2, *rw3;
  if (n_in >= 29) {
    const float* const* p = (const float* const*)(d_in + 1);
    g1W = p[0];  g1ad = p[1];  g1as = p[2];  g1b = p[3];
    g2W = p[4];  g2ad = p[5];  g2as = p[6];  g2b = p[7];
    rb1 = p[8];  rb2 = p[9];  rb3 = p[10];
    s1b1 = p[11]; s1b2 = p[12]; s1w1 = p[13]; s1w2 = p[14];
    s2b1 = p[15]; s2b2 = p[16]; s2w1 = p[17]; s2w2 = p[18];
    s3b1 = p[19]; s3b2 = p[20]; s3w1 = p[21]; s3w2 = p[22];
    rw1 = p[23]; rw2 = p[24]; rw3 = p[25];
  } else {
    const float* P = (const float*)d_in[1];
    g1W = P + 0;      g1ad = P + 2048;  g1as = P + 2112;  g1b = P + 2176;
    g2W = P + 2240;   g2ad = P + 4288;  g2as = P + 4320;  g2b = P + 4352;
    rb1 = P + 4384;   rb2 = P + 4640;   rb3 = P + 4768;
    s1b1 = P + 4832;  s1b2 = P + 4848;  s1w1 = P + 5104;  s1w2 = P + 9200;
    s2b1 = P + 13296; s2b2 = P + 13304; s2w1 = P + 13432; s2w2 = P + 14456;
    s3b1 = P + 15480; s3b2 = P + 15484; s3w1 = P + 15548; s3w2 = P + 15804;
    rw1 = P + 16060;  rw2 = P + 147132; rw3 = P + 179900;
  }

  // Workspace bump allocator
  char* w = (char*)d_ws;
  size_t off = 0;
  auto alloc = [&](size_t bytes) { void* q = w + off; off += alup(bytes); return q; };

  unsigned short* A1bf  = (unsigned short*)alloc((size_t)4096 * 512 * 2);
  unsigned short* w1bf  = (unsigned short*)alloc((size_t)256 * 512 * 2);
  unsigned short* w2bf  = (unsigned short*)alloc((size_t)128 * 256 * 2);
  unsigned short* w3bf  = (unsigned short*)alloc((size_t)64 * 128 * 2);
  unsigned short* W1tbf = (unsigned short*)alloc((size_t)64 * 32 * 2);
  unsigned short* W2tbf = (unsigned short*)alloc((size_t)32 * 64 * 2);
  float* Y1 = (float*)alloc((size_t)4096 * 256 * 4);
  float* Y2 = (float*)alloc((size_t)4096 * 128 * 4);
  float* Y3 = (float*)alloc((size_t)4096 * 64 * 4);
  unsigned short* X2bf = (unsigned short*)alloc((size_t)4096 * 256 * 2);
  unsigned short* X3bf = (unsigned short*)alloc((size_t)4096 * 128 * 2);
  unsigned short* Xgbf = (unsigned short*)alloc((size_t)8192 * 32 * 2);
  float* mean1 = (float*)alloc(256 * 4);
  float* mean2 = (float*)alloc(128 * 4);
  float* mean3 = (float*)alloc(64 * 4);
  float* gate1 = (float*)alloc(256 * 4);
  float* gate2 = (float*)alloc(128 * 4);
  float* gate3 = (float*)alloc(64 * 4);
  float* H1    = (float*)alloc((size_t)8192 * 64 * 4);
  float* as1   = (float*)alloc((size_t)8192 * 4 * 4);
  float* ad1   = (float*)alloc((size_t)8192 * 4 * 4);
  float* X1out = (float*)alloc((size_t)8192 * 64 * 4);
  unsigned short* X2gbf = (unsigned short*)alloc((size_t)8192 * 64 * 2);
  float* H2    = (float*)alloc((size_t)8192 * 32 * 4);
  float* as2   = (float*)alloc((size_t)8192 * 4 * 4);
  float* ad2   = (float*)alloc((size_t)8192 * 4 * 4);

  const int T = 256;
  auto nb = [](int n, int t) { return (n + t - 1) / t; };

  // ---- precompute bf16 operands -----------------------------------------
  k_transpose_cvt<<<nb(512 * 4096, T), T, 0, stream>>>(x2_raw, A1bf, 512, 4096);
  k_cvt_bf<<<nb(131072, T), T, 0, stream>>>(rw1, w1bf, 131072);
  k_cvt_bf<<<nb(32768, T), T, 0, stream>>>(rw2, w2bf, 32768);
  k_cvt_bf<<<nb(8192, T), T, 0, stream>>>(rw3, w3bf, 8192);
  k_transpose_cvt<<<nb(2048, T), T, 0, stream>>>(g1W, W1tbf, 32, 64);  // [32][64] -> [64][32]
  k_transpose_cvt<<<nb(2048, T), T, 0, stream>>>(g2W, W2tbf, 64, 32);  // [64][32] -> [32][64]

  // ---- ChannelReducer stage 1: 512 -> 256 (M=4096) ----------------------
  // tiles = (4096/16) * (256/64) = 1024; 8 waves/block -> 128 blocks
  k_gemm_bf16<4><<<128, T, 0, stream>>>(A1bf, w1bf, rb1, Y1, 4096, 256, 512, 1);
  k_col_mean<<<256, T, 0, stream>>>(Y1, mean1, 4096, 256);
  k_se_mlp<<<1, T, 0, stream>>>(mean1, s1w1, s1b1, s1w2, s1b2, gate1, 256, 16);
  k_scale_cvt<<<nb(4096 * 256, T), T, 0, stream>>>(Y1, gate1, X2bf, 4096, 256);

  // ---- stage 2: 256 -> 128 ----------------------------------------------
  k_gemm_bf16<4><<<64, T, 0, stream>>>(X2bf, w2bf, rb2, Y2, 4096, 128, 256, 1);  // 512 tiles
  k_col_mean<<<128, T, 0, stream>>>(Y2, mean2, 4096, 128);
  k_se_mlp<<<1, T, 0, stream>>>(mean2, s2w1, s2b1, s2w2, s2b2, gate2, 128, 8);
  k_scale_cvt<<<nb(4096 * 128, T), T, 0, stream>>>(Y2, gate2, X3bf, 4096, 128);

  // ---- stage 3: 128 -> 64 -----------------------------------------------
  k_gemm_bf16<4><<<32, T, 0, stream>>>(X3bf, w3bf, rb3, Y3, 4096, 64, 128, 1);   // 256 tiles
  k_col_mean<<<64, T, 0, stream>>>(Y3, mean3, 4096, 64);
  k_se_mlp<<<1, T, 0, stream>>>(mean3, s3w1, s3b1, s3w2, s3b2, gate3, 64, 4);
  k_stage3_nodes<<<nb(262144, T), T, 0, stream>>>(Y3, gate3, Xgbf, 262144);

  // ---- GAT layer 1: 32 -> 4 heads x 16 ----------------------------------
  k_gemm_bf16<4><<<64, T, 0, stream>>>(Xgbf, W1tbf, nullptr, H1, 8192, 64, 32, 0); // 512 tiles
  k_gat_scores<16><<<nb(8192, T), T, 0, stream>>>(H1, g1as, g1ad, as1, ad1);
  k_gat_aggregate<16><<<1024, T, 0, stream>>>(H1, as1, ad1, g1b, X1out, X2gbf);

  // ---- GAT layer 2: 64 -> 4 heads x 8 -> d_out --------------------------
  k_gemm_bf16<2><<<64, T, 0, stream>>>(X2gbf, W2tbf, nullptr, H2, 8192, 32, 64, 0); // 512 tiles
  k_gat_scores<8><<<nb(8192, T), T, 0, stream>>>(H2, g2as, g2ad, as2, ad2);
  k_gat_aggregate<8><<<1024, T, 0, stream>>>(H2, as2, ad2, g2b, (float*)d_out, nullptr);
}
